// SimpleSAE_30485677867296
// MI455X (gfx1250) — matrix operand order repaired and checked by reading, not compiled, not run on hardware
//
#include <hip/hip_runtime.h>
#include <hip/hip_bf16.h>

typedef __attribute__((ext_vector_type(16))) _Float16 v16h;
typedef __attribute__((ext_vector_type(8)))  _Float16 v8h;
typedef __attribute__((ext_vector_type(8)))  float    v8f;

#define N_ROWS   8192
#define D_MODEL  768
#define D_SAE    16384
#define K_SEL    32

// ---------------------------------------------------------------------------
// Kernel 1: W_dec transpose  [768, 16384] -> [16384, 768]  (into d_ws)
// ---------------------------------------------------------------------------
__global__ void sae_transpose_wdec(const float* __restrict__ Wdec,
                                   float* __restrict__ WdT) {
    __shared__ float tile[32][33];
    const int bs = blockIdx.x * 32;          // s-dim base (16384)
    const int bd = blockIdx.y * 32;          // d-dim base (768)
    const int tx = threadIdx.x & 31;
    const int ty = threadIdx.x >> 5;         // 0..7
    #pragma unroll
    for (int r = 0; r < 32; r += 8)
        tile[r + ty][tx] = Wdec[(size_t)(bd + r + ty) * D_SAE + bs + tx];
    __syncthreads();
    #pragma unroll
    for (int r = 0; r < 32; r += 8)
        WdT[(size_t)(bs + r + ty) * D_MODEL + bd + tx] = tile[tx][r + ty];
}

// ---------------------------------------------------------------------------
// Kernel 2: encoder GEMM  pre[n,s] = sum_d x[n,d]*W_enc[s,d] + b_enc[s]
// f32 inputs converted to f16 during LDS staging; f32 WMMA accumulation.
// Block: 256 thr (8 waves). Tile: 128(M) x 128(N), K-step 32.
// Wave (wm in 0..1, wn in 0..3) computes 64x32 = 4x2 WMMA frags.
// ---------------------------------------------------------------------------
#define LDS_PITCH 40   // f16 elements per row (padded: 80B -> conflict-free)

__device__ __forceinline__ v16h load_frag_A(const _Float16* lds, int row, int khalf) {
    // ISA A layout: lane holds K = {8kh..8kh+7} and {16+8kh..16+8kh+7}
    const _Float16* p = lds + row * LDS_PITCH + khalf * 8;
    v8h lo = *(const v8h*)p;
    v8h hi = *(const v8h*)(p + 16);
    return __builtin_shufflevector(lo, hi, 0,1,2,3,4,5,6,7,8,9,10,11,12,13,14,15);
}

__device__ __forceinline__ v16h load_frag_B(const _Float16* lds, int col, int khalf) {
    // ISA B layout: lane holds K = {8kh..8kh+7} and {16+8kh..16+8kh+7}
    const _Float16* p = lds + col * LDS_PITCH + khalf * 8;
    v8h lo = *(const v8h*)p;
    v8h hi = *(const v8h*)(p + 16);
    return __builtin_shufflevector(lo, hi, 0,1,2,3,4,5,6,7,8,9,10,11,12,13,14,15);
}

__global__ void __launch_bounds__(256)
sae_encoder_gemm(const float* __restrict__ x,
                 const float* __restrict__ W_enc,
                 const float* __restrict__ b_enc,
                 float* __restrict__ pre) {
    __shared__ _Float16 sA[128 * LDS_PITCH];
    __shared__ _Float16 sB[128 * LDS_PITCH];

    const int tid  = threadIdx.x;
    const int lane = tid & 31;
    const int wv   = tid >> 5;          // 0..7
    const int wm   = wv >> 2;           // 0..1  -> M offset wm*64
    const int wn   = wv & 3;            // 0..3  -> N offset wn*32
    const int lrow = lane & 15;
    const int kh   = lane >> 4;

    const int n0 = blockIdx.x * 128;    // d_sae tile base
    const int m0 = blockIdx.y * 128;    // row tile base

    v8f acc[4][2] = {};

    const int srow = tid >> 1;          // 0..127
    const int shalf = tid & 1;          // which 16-float chunk of the 32-K step

    for (int kb = 0; kb < D_MODEL; kb += 32) {
        // ---- stage x & W_enc tiles to LDS with f32 -> f16 convert ----
        {
            const float* xp = x     + (size_t)(m0 + srow) * D_MODEL + kb + shalf * 16;
            const float* wp = W_enc + (size_t)(n0 + srow) * D_MODEL + kb + shalf * 16;
            _Float16* da = sA + srow * LDS_PITCH + shalf * 16;
            _Float16* db = sB + srow * LDS_PITCH + shalf * 16;
            #pragma unroll
            for (int c = 0; c < 4; ++c) {
                float4 vx = ((const float4*)xp)[c];
                float4 vw = ((const float4*)wp)[c];
                da[4*c + 0] = (_Float16)vx.x; da[4*c + 1] = (_Float16)vx.y;
                da[4*c + 2] = (_Float16)vx.z; da[4*c + 3] = (_Float16)vx.w;
                db[4*c + 0] = (_Float16)vw.x; db[4*c + 1] = (_Float16)vw.y;
                db[4*c + 2] = (_Float16)vw.z; db[4*c + 3] = (_Float16)vw.w;
            }
        }
        __syncthreads();

        // ---- fragments + WMMA ----
        v16h af[4], bf[2];
        #pragma unroll
        for (int i = 0; i < 4; ++i)
            af[i] = load_frag_A(sA, wm * 64 + i * 16 + lrow, kh);
        #pragma unroll
        for (int j = 0; j < 2; ++j)
            bf[j] = load_frag_B(sB, wn * 32 + j * 16 + lrow, kh);

        #pragma unroll
        for (int i = 0; i < 4; ++i)
            #pragma unroll
            for (int j = 0; j < 2; ++j)
                acc[i][j] = __builtin_amdgcn_wmma_f32_16x16x32_f16(
                    false, af[i], false, bf[j], (short)0, acc[i][j], false, false);
        __syncthreads();
    }

    // ---- epilogue: add bias, store pre_activation ----
    #pragma unroll
    for (int j = 0; j < 2; ++j) {
        const int col  = n0 + wn * 32 + j * 16 + lrow;
        const float bias = b_enc[col];
        #pragma unroll
        for (int i = 0; i < 4; ++i) {
            const int rbase = m0 + wm * 64 + i * 16 + kh * 8;
            #pragma unroll
            for (int r = 0; r < 8; ++r)
                pre[(size_t)(rbase + r) * D_SAE + col] = acc[i][j][r] + bias;
        }
    }
}

// ---------------------------------------------------------------------------
// Kernel 3: per-row top-k (k=32) + latents scatter + sparse decode
// One block (256 thr) per row; 64 values/thread in registers; iterative
// argmax via wave32 shfl_xor + 8-wave LDS reduce; 64-bit selected mask.
// ---------------------------------------------------------------------------
__global__ void __launch_bounds__(256)
sae_topk_decode(const float* __restrict__ pre,
                const float* __restrict__ Wdec,
                const float* __restrict__ WdT,
                float* __restrict__ lat,
                float* __restrict__ rec,
                int useT) {
    __shared__ float s_wv[8];
    __shared__ int   s_wi[8];
    __shared__ float topv[K_SEL];
    __shared__ int   topi[K_SEL];
    __shared__ int   s_win;

    const int row = blockIdx.x;
    const int tid = threadIdx.x;
    const size_t rbase = (size_t)row * D_SAE;

    float v[64];
    #pragma unroll
    for (int i = 0; i < 64; ++i)
        v[i] = pre[rbase + (size_t)i * 256 + tid];

    unsigned long long mask = 0ull;
    const float NEG = -__builtin_inff();

    for (int it = 0; it < K_SEL; ++it) {
        // local argmax over un-selected elements
        float best = NEG;
        int   bi   = 0;
        #pragma unroll
        for (int i = 0; i < 64; ++i) {
            const float cand = ((mask >> i) & 1ull) ? NEG : v[i];
            if (cand > best) { best = cand; bi = i; }
        }
        int gidx = bi * 256 + tid;               // global s index

        // wave32 reduction
        #pragma unroll
        for (int off = 16; off > 0; off >>= 1) {
            const float ov = __shfl_xor(best, off, 32);
            const int   og = __shfl_xor(gidx, off, 32);
            if (ov > best) { best = ov; gidx = og; }
        }
        if ((tid & 31) == 0) { s_wv[tid >> 5] = best; s_wi[tid >> 5] = gidx; }
        __syncthreads();
        if (tid == 0) {
            float bb = s_wv[0]; int bg = s_wi[0];
            #pragma unroll
            for (int w = 1; w < 8; ++w)
                if (s_wv[w] > bb) { bb = s_wv[w]; bg = s_wi[w]; }
            topv[it] = bb; topi[it] = bg; s_win = bg;
        }
        __syncthreads();
        const int win = s_win;
        if ((win & 255) == tid) mask |= (1ull << (win >> 8));
        // next iteration's first __syncthreads guards s_win reuse
    }
    __syncthreads();

    // latents: scatter top-k values, zeros elsewhere (full dense row write)
    #pragma unroll
    for (int i = 0; i < 64; ++i)
        lat[rbase + (size_t)i * 256 + tid] = ((mask >> i) & 1ull) ? v[i] : 0.0f;

    // decode: rec[row,d] = sum_j topv[j] * W_dec[d, topi[j]]
    if (useT) {
        for (int d = tid; d < D_MODEL; d += 256) {
            float acc = 0.0f;
            #pragma unroll
            for (int j = 0; j < K_SEL; ++j)
                acc = fmaf(topv[j], WdT[(size_t)topi[j] * D_MODEL + d], acc);
            rec[(size_t)row * D_MODEL + d] = acc;
        }
    } else {
        for (int d = tid; d < D_MODEL; d += 256) {
            float acc = 0.0f;
            #pragma unroll
            for (int j = 0; j < K_SEL; ++j)
                acc = fmaf(topv[j], Wdec[(size_t)d * D_SAE + topi[j]], acc);
            rec[(size_t)row * D_MODEL + d] = acc;
        }
    }
}

// ---------------------------------------------------------------------------
extern "C" void kernel_launch(void* const* d_in, const int* in_sizes, int n_in,
                              void* d_out, int out_size, void* d_ws, size_t ws_size,
                              hipStream_t stream) {
    const float* x     = (const float*)d_in[0];
    const float* W_enc = (const float*)d_in[1];
    const float* b_enc = (const float*)d_in[2];
    const float* W_dec = (const float*)d_in[3];
    // d_in[4] = k (always 32 for this problem)

    float* rec = (float*)d_out;                                   // [8192, 768]
    float* lat = rec + (size_t)N_ROWS * D_MODEL;                  // [8192, 16384]
    float* pre = lat + (size_t)N_ROWS * D_SAE;                    // [8192, 16384]

    const size_t wdt_bytes = (size_t)D_SAE * D_MODEL * sizeof(float);
    const int useT = (ws_size >= wdt_bytes) ? 1 : 0;
    float* WdT = (float*)d_ws;

    if (useT) {
        sae_transpose_wdec<<<dim3(D_SAE / 32, D_MODEL / 32), 256, 0, stream>>>(W_dec, WdT);
    }
    sae_encoder_gemm<<<dim3(D_SAE / 128, N_ROWS / 128), 256, 0, stream>>>(
        x, W_enc, b_enc, pre);
    sae_topk_decode<<<N_ROWS, 256, 0, stream>>>(pre, W_dec, WdT, lat, rec, useT);
}
